// Decoder_trans_781684047977
// MI455X (gfx1250) — matrix operand order repaired and checked by reading, not compiled, not run on hardware
//
#include <hip/hip_runtime.h>
#include <hip/hip_bf16.h>

// ---------------------------------------------------------------------------
// Transformer decoder (B=4, S=1024, H=1024, NH=16, HD=64, PF=4096, L=6).
// bf16 operand pipeline: weights converted+transposed to bf16 [N,K] once,
// every GEMM consumer reads bf16 produced by the upstream kernel's epilogue.
// GEMMs: v_wmma_f32_16x16x32_bf16, double-buffered LDS staging via
// GLOBAL_LOAD_ASYNC_TO_LDS_B128 (ASYNCcnt) when the builtin is available.
// ---------------------------------------------------------------------------

typedef __attribute__((ext_vector_type(16))) __bf16 v16bf;
typedef __attribute__((ext_vector_type(8)))  float  v8f;
typedef __attribute__((__vector_size__(16))) int    v4i;

union Frag16 { v16bf v; unsigned int u32[8]; };

constexpr int Bc  = 4;
constexpr int SQc = 1024;
constexpr int SKc = 1024;
constexpr int INc = 128;
constexpr int Hc  = 1024;
constexpr int NHc = 16;
constexpr int PFc = 4096;
constexpr int Lc  = 6;
constexpr int HDc = 64;
constexpr float SCALEc = 32.0f;   // sqrt(H)

__device__ __forceinline__ unsigned short f2bf(float f) {
  unsigned int u = __float_as_uint(f);
  u += 0x7FFFu + ((u >> 16) & 1u);          // round-to-nearest-even
  return (unsigned short)(u >> 16);
}

// ---- async-to-LDS staging (gfx1250 ASYNCcnt path) with safe fallback ------
#if __has_builtin(__builtin_amdgcn_global_load_async_to_lds_b128) && \
    __has_builtin(__builtin_amdgcn_s_wait_asynccnt)
#define ASYNC_STAGE 1
#else
#define ASYNC_STAGE 0
#endif

__device__ __forceinline__ void stage16(const unsigned short* g, unsigned short* l) {
#if ASYNC_STAGE
  __builtin_amdgcn_global_load_async_to_lds_b128(
      (__attribute__((address_space(1))) v4i*)g,
      (__attribute__((address_space(3))) v4i*)l, 0, 0);
#else
  *(uint4*)l = *(const uint4*)g;
#endif
}
__device__ __forceinline__ void stage_wait() {
#if ASYNC_STAGE
  __builtin_amdgcn_s_wait_asynccnt(0);
#endif
}

// epilogue modes
enum { M_EMBED = 0, M_QKV = 1, M_VT = 2, M_MERGE = 3, M_SCORES = 4,
       M_PLAINF = 5, M_RELUB = 6 };

// ---------------------------------------------------------------------------
// WMMA GEMM, bf16 operands: A[M,K] row-major, B[N,K] row-major (pre-
// transposed). Block tile 128x64, BK=32, 8 waves x (32x32 acc). Double-
// buffered LDS. M,N,K assumed multiples of 128/64/32 (true for all calls).
// ---------------------------------------------------------------------------
template <int MODE>
__global__ __launch_bounds__(256)
void gemm_wmma(const unsigned short* __restrict__ A,
               const unsigned short* __restrict__ Bm,
               float* __restrict__ C, unsigned short* __restrict__ Cb,
               const float* __restrict__ bias, const float* __restrict__ aux,
               int K, int lda, int ldb, int ldc, float alpha,
               long sA, long sB, long sC) {
  constexpr int BM = 128, BN = 64, BK = 32;
  constexpr int LDSA = BK + 8;   // 40 ushorts = 80B rows: 16B aligned, 20-bank stride
  __shared__ __align__(16) unsigned short As[2][BM * LDSA];
  __shared__ __align__(16) unsigned short Bs[2][BN * LDSA];

  const int tid = threadIdx.x;
  const int z   = blockIdx.z;
  A  += (long)z * sA;
  Bm += (long)z * sB;

  const int rowBase = blockIdx.y * BM;
  const int colBase = blockIdx.x * BN;

  const int wave  = tid >> 5;
  const int wm    = wave >> 1;    // 0..3 : 32-row strip
  const int wn    = wave & 1;     // 0..1 : 32-col strip
  const int lane  = tid & 31;
  const int l15   = lane & 15;
  const int lhalf = lane >> 4;

  // staging: A tile 128x32 ushorts = 512 x 16B (2/thread), B tile 64x32 = 256 (1/thread)
  const int arow0 = tid >> 2, aseg = tid & 3;
  auto stageA = [&](int buf, int kb) {
#pragma unroll
    for (int i = 0; i < 2; ++i) {
      const int row = arow0 + i * 64;
      stage16(A + (long)(rowBase + row) * lda + kb + aseg * 8,
              &As[buf][row * LDSA + aseg * 8]);
    }
  };
  auto stageB = [&](int buf, int kb) {
    stage16(Bm + (long)(colBase + arow0) * ldb + kb + aseg * 8,
            &Bs[buf][arow0 * LDSA + aseg * 8]);
  };

  v8f acc[2][2];
#pragma unroll
  for (int ti = 0; ti < 2; ++ti)
#pragma unroll
    for (int tj = 0; tj < 2; ++tj)
#pragma unroll
      for (int i = 0; i < 8; ++i) acc[ti][tj][i] = 0.0f;

  const int nk = K / BK;
  stageA(0, 0);
  stageB(0, 0);

  for (int ik = 0; ik < nk; ++ik) {
    const int bsel = ik & 1;
    stage_wait();
    __syncthreads();
    if (ik + 1 < nk) {             // prefetch next tile into the other buffer
      stageA(bsel ^ 1, (ik + 1) * BK);
      stageB(bsel ^ 1, (ik + 1) * BK);
    }

    Frag16 a0, a1, b0, b1;
    {
      const int m0 = wm * 32 + l15;
      const unsigned short* as = &As[bsel][0];
#pragma unroll
      for (int j = 0; j < 8; ++j) {
        // A layout (matches B): lanes 0-15 K=0..15, lanes 16-31 K=16..31
        const int kk = (lhalf << 4) + (j << 1);
        a0.u32[j] = *(const unsigned int*)&as[m0 * LDSA + kk];
        a1.u32[j] = *(const unsigned int*)&as[(m0 + 16) * LDSA + kk];
      }
      const int n0 = wn * 32 + l15;
      const unsigned short* bs = &Bs[bsel][0];
#pragma unroll
      for (int j = 0; j < 8; ++j) {
        // B layout (matches A): lanes 0-15 K=0..15, lanes 16-31 K=16..31
        const int kk = (lhalf << 4) + (j << 1);
        b0.u32[j] = *(const unsigned int*)&bs[n0 * LDSA + kk];
        b1.u32[j] = *(const unsigned int*)&bs[(n0 + 16) * LDSA + kk];
      }
    }
    acc[0][0] = __builtin_amdgcn_wmma_f32_16x16x32_bf16(false, a0.v, false, b0.v,
                                                        (short)0, acc[0][0], false, false);
    acc[0][1] = __builtin_amdgcn_wmma_f32_16x16x32_bf16(false, a0.v, false, b1.v,
                                                        (short)0, acc[0][1], false, false);
    acc[1][0] = __builtin_amdgcn_wmma_f32_16x16x32_bf16(false, a1.v, false, b0.v,
                                                        (short)0, acc[1][0], false, false);
    acc[1][1] = __builtin_amdgcn_wmma_f32_16x16x32_bf16(false, a1.v, false, b1.v,
                                                        (short)0, acc[1][1], false, false);
    __syncthreads();
  }

  // ---- epilogue: lane -> (half,n), VGPR i -> row 8*half+i -----------------
#pragma unroll
  for (int ti = 0; ti < 2; ++ti) {
#pragma unroll
    for (int tj = 0; tj < 2; ++tj) {
      const int gn  = colBase + wn * 32 + tj * 16 + l15;
      const int gm0 = rowBase + wm * 32 + ti * 16 + (lhalf << 3);
#pragma unroll
      for (int i = 0; i < 8; ++i) {
        const int gm = gm0 + i;
        float v = acc[ti][tj][i];
        if constexpr (MODE == M_EMBED) {
          v = (v + bias[gn]) * alpha + aux[(long)(gm & (SQc - 1)) * Hc + gn];
          C[(long)gm * ldc + gn]  = v;
          Cb[(long)gm * ldc + gn] = f2bf(v);
        } else if constexpr (MODE == M_QKV) {
          v += bias[gn];
          const int b = gm >> 10, s = gm & 1023;   // SQ == 1024
          const int h = gn >> 6,  d = gn & 63;     // HD == 64
          Cb[((long)((b * NHc + h) * SQc + s)) * HDc + d] = f2bf(v);
        } else if constexpr (MODE == M_VT) {       // head-transposed V [B,NH,HD,SQ]
          v += bias[gn];
          const int b = gm >> 10, s = gm & 1023;
          const int h = gn >> 6,  d = gn & 63;
          Cb[((long)((b * NHc + h) * HDc + d)) * SQc + s] = f2bf(v);
        } else if constexpr (MODE == M_MERGE) {    // heads merged [B,S,H]
          const int b = z >> 4, h = z & 15;        // NH == 16
          Cb[((long)(b * SQc + gm)) * Hc + h * HDc + gn] = f2bf(v);
        } else if constexpr (MODE == M_SCORES) {   // batched fp32 scores
          C[(long)z * sC + (long)gm * ldc + gn] = v * alpha;
        } else if constexpr (MODE == M_PLAINF) {   // fp32 + bias
          C[(long)gm * ldc + gn] = v + bias[gn];
        } else {                                   // M_RELUB: bf16 + bias + relu
          Cb[(long)gm * ldc + gn] = f2bf(fmaxf(v + bias[gn], 0.0f));
        }
      }
    }
  }
}

// ---------------------------------------------------------------------------
// fp32 [R,C] -> bf16 [C,R] transpose-convert (weights), batched over z.
// ---------------------------------------------------------------------------
__global__ __launch_bounds__(256)
void transpose_cvt_kernel(const float* __restrict__ src, unsigned short* __restrict__ dst,
                          int R, int C, long sSrc, long sDst) {
  __shared__ float t[32][33];
  src += (long)blockIdx.z * sSrc;
  dst += (long)blockIdx.z * sDst;
  const int c0 = blockIdx.x * 32, r0 = blockIdx.y * 32;
  const int tx = threadIdx.x & 31, ty = threadIdx.x >> 5;   // 32 x 8
#pragma unroll
  for (int i = 0; i < 4; ++i)
    t[ty + i * 8][tx] = src[(long)(r0 + ty + i * 8) * C + c0 + tx];
  __syncthreads();
#pragma unroll
  for (int i = 0; i < 4; ++i) {
    const int cc = ty + i * 8;
    dst[(long)(c0 + cc) * R + r0 + tx] = f2bf(t[tx][cc]);
  }
}

// fp32 -> bf16 elementwise (x4 vectorized); n must be a multiple of 4.
__global__ __launch_bounds__(256)
void cvt_bf16_kernel(const float* __restrict__ src, unsigned short* __restrict__ dst, long n4) {
  const long i = (long)blockIdx.x * 256 + threadIdx.x;
  if (i >= n4) return;
  const float4 f = ((const float4*)src)[i];
  ushort4 o;
  o.x = f2bf(f.x); o.y = f2bf(f.y); o.z = f2bf(f.z); o.w = f2bf(f.w);
  ((ushort4*)dst)[i] = o;
}

// ---------------------------------------------------------------------------
// Masked softmax over SK=1024 columns. grid=(SQ, B*NH), 256 threads.
// MM==0: 2D causal mask [SQ,SK]; MM==1: key mask [B,SK].
// Writes bf16 probabilities (attnV operand) + optional fp32 mirror (d_out).
// ---------------------------------------------------------------------------
template <int MM>
__global__ __launch_bounds__(256)
void softmax_kernel(const float* __restrict__ scores, const int* __restrict__ mask,
                    unsigned short* __restrict__ probs, float* __restrict__ mirror) {
  __shared__ float red[256];
  const int q = blockIdx.x;
  const int z = blockIdx.y;
  const int tid = threadIdx.x;
  const long rowoff = ((long)z * SQc + q) * (long)SKc;
  const float* row = scores + rowoff;
  const int b = z >> 4;

  float vals[4];
  bool  ok[4];
  float m = -3.0e38f;
#pragma unroll
  for (int i = 0; i < 4; ++i) {
    const int col = tid + i * 256;
    ok[i]   = (MM == 0) ? (mask[q * SKc + col] != 0) : (mask[b * SKc + col] != 0);
    vals[i] = row[col];
    if (ok[i]) m = fmaxf(m, vals[i]);
  }
  red[tid] = m;
  __syncthreads();
  for (int s = 128; s > 0; s >>= 1) {
    if (tid < s) red[tid] = fmaxf(red[tid], red[tid + s]);
    __syncthreads();
  }
  m = red[0];
  __syncthreads();

  float sum = 0.0f;
#pragma unroll
  for (int i = 0; i < 4; ++i) {
    vals[i] = ok[i] ? __expf(vals[i] - m) : 0.0f;  // masked -> exactly 0 (== -1e10 path)
    sum += vals[i];
  }
  red[tid] = sum;
  __syncthreads();
  for (int s = 128; s > 0; s >>= 1) {
    if (tid < s) red[tid] += red[tid + s];
    __syncthreads();
  }
  const float inv = 1.0f / red[0];
#pragma unroll
  for (int i = 0; i < 4; ++i) {
    const int col = tid + i * 256;
    const float p = vals[i] * inv;
    probs[rowoff + col] = f2bf(p);
    if (mirror) mirror[rowoff + col] = p;
  }
}

// ---------------------------------------------------------------------------
// out = LayerNorm(x + t)*g + b ; writes fp32 (residual) + bf16 (GEMM operand).
// ---------------------------------------------------------------------------
__global__ __launch_bounds__(256)
void add_ln_kernel(const float* __restrict__ x, const float* __restrict__ t,
                   const float* __restrict__ g, const float* __restrict__ be,
                   float* __restrict__ out, unsigned short* __restrict__ obf) {
  __shared__ float red[256];
  const long row = blockIdx.x;
  const int tid = threadIdx.x;
  const float* xr = x + row * Hc;
  const float* tr = t + row * Hc;

  float v[4];
  float s = 0.0f;
#pragma unroll
  for (int i = 0; i < 4; ++i) {
    const int idx = tid + i * 256;
    v[i] = xr[idx] + tr[idx];
    s += v[i];
  }
  red[tid] = s;
  __syncthreads();
  for (int st = 128; st > 0; st >>= 1) {
    if (tid < st) red[tid] += red[tid + st];
    __syncthreads();
  }
  const float mu = red[0] * (1.0f / Hc);
  __syncthreads();

  float vs = 0.0f;
#pragma unroll
  for (int i = 0; i < 4; ++i) { const float d = v[i] - mu; vs += d * d; }
  red[tid] = vs;
  __syncthreads();
  for (int st = 128; st > 0; st >>= 1) {
    if (tid < st) red[tid] += red[tid + st];
    __syncthreads();
  }
  const float rs = rsqrtf(red[0] * (1.0f / Hc) + 1e-5f);
#pragma unroll
  for (int i = 0; i < 4; ++i) {
    const int idx = tid + i * 256;
    const float o = (v[i] - mu) * rs * g[idx] + be[idx];
    out[row * Hc + idx] = o;
    obf[row * Hc + idx] = f2bf(o);
  }
}

// ---------------------------------------------------------------------------
extern "C" void kernel_launch(void* const* d_in, const int* in_sizes, int n_in,
                              void* d_out, int out_size, void* d_ws, size_t ws_size,
                              hipStream_t stream) {
  (void)in_sizes; (void)n_in; (void)out_size; (void)ws_size;
  typedef unsigned short u16;

  const float* dec_x    = (const float*)d_in[0];
  const float* enc_x    = (const float*)d_in[1];
  const int*   dec_mask = (const int*)d_in[2];
  const int*   enc_mask = (const int*)d_in[3];
  const float* emb_w    = (const float*)d_in[4];
  const float* emb_b    = (const float*)d_in[5];
  const float* pos_e    = (const float*)d_in[6];
  const float* attn_w   = (const float*)d_in[7];
  const float* attn_b   = (const float*)d_in[8];
  const float* ln_g     = (const float*)d_in[9];
  const float* ln_b     = (const float*)d_in[10];
  const float* ff_w1    = (const float*)d_in[11];
  const float* ff_b1    = (const float*)d_in[12];
  const float* ff_w2    = (const float*)d_in[13];
  const float* ff_b2    = (const float*)d_in[14];
  const float* out_w    = (const float*)d_in[15];
  const float* out_b    = (const float*)d_in[16];

  float* out      = (float*)d_out;
  float* attn_out = out + (long)Bc * SQc * INc;     // fp32 probs [B,NH,SQ,SK]

  const long MT = (long)Bc * SQc;                   // 4096 token rows
  // ---- workspace carve: fp32 first, then bf16 regions ----
  float* wf = (float*)d_ws;
  float* trgA   = wf; wf += MT * Hc;
  float* trgB   = wf; wf += MT * Hc;
  float* delta  = wf; wf += MT * Hc;                // o-proj / ffn2 output
  float* scores = wf; wf += (long)Bc * NHc * SQc * SKc;

  u16* wb = (u16*)wf;
  u16* trg_bf   = wb; wb += MT * Hc;
  u16* tmp_bf   = wb; wb += MT * Hc;                // merged attn output
  u16* q_bf     = wb; wb += MT * Hc;
  u16* k_bf     = wb; wb += MT * Hc;
  u16* vT_bf    = wb; wb += MT * Hc;                // [B,NH,HD,SQ]
  u16* enc_bf   = wb; wb += MT * Hc;
  u16* dec_bf   = wb; wb += MT * INc;
  u16* probs_bf = wb; wb += (long)Bc * NHc * SQc * SKc;
  u16* ffnmid_bf = probs_bf;                        // aliases probs (disjoint in time)
  u16* wAttnT   = wb; wb += (long)Lc * 2 * 4 * Hc * Hc;   // 48 x [H,H]
  u16* wFF1T    = wb; wb += (long)Lc * PFc * Hc;          // 6 x [PF,H]
  u16* wFF2T    = wb; wb += (long)Lc * Hc * PFc;          // 6 x [H,PF]
  u16* wEmbT    = wb; wb += (long)Hc * INc;               // [H,IN]
  u16* wOutT    = wb; wb += (long)INc * Hc;               // [IN,H]

  const dim3 blk(256);
  auto grd = [](long Mi, long Ni, int Zi) {
    return dim3((unsigned)(Ni / 64), (unsigned)(Mi / 128), (unsigned)Zi);
  };

  // ---- one-time conversions (bf16 operand pipeline) ----
  cvt_bf16_kernel<<<dim3((unsigned)((MT * Hc / 4 + 255) / 256)), blk, 0, stream>>>(
      enc_x, enc_bf, MT * Hc / 4);
  cvt_bf16_kernel<<<dim3((unsigned)((MT * INc / 4 + 255) / 256)), blk, 0, stream>>>(
      dec_x, dec_bf, MT * INc / 4);
  transpose_cvt_kernel<<<dim3(Hc / 32, Hc / 32, Lc * 2 * 4), blk, 0, stream>>>(
      attn_w, wAttnT, Hc, Hc, (long)Hc * Hc, (long)Hc * Hc);
  transpose_cvt_kernel<<<dim3(PFc / 32, Hc / 32, Lc), blk, 0, stream>>>(
      ff_w1, wFF1T, Hc, PFc, (long)Hc * PFc, (long)PFc * Hc);
  transpose_cvt_kernel<<<dim3(Hc / 32, PFc / 32, Lc), blk, 0, stream>>>(
      ff_w2, wFF2T, PFc, Hc, (long)PFc * Hc, (long)Hc * PFc);
  transpose_cvt_kernel<<<dim3(Hc / 32, INc / 32, 1), blk, 0, stream>>>(
      emb_w, wEmbT, INc, Hc, 0, 0);
  transpose_cvt_kernel<<<dim3(INc / 32, Hc / 32, 1), blk, 0, stream>>>(
      out_w, wOutT, Hc, INc, 0, 0);

  // trg = (dec_x @ emb_w + emb_b)*sqrt(H) + pos_e  (fp32 + bf16 outputs)
  gemm_wmma<M_EMBED><<<grd(MT, Hc, 1), blk, 0, stream>>>(
      dec_bf, wEmbT, trgA, trg_bf, emb_b, pos_e, INc, INc, INc, Hc, SCALEc, 0, 0, 0);

  float* cur = trgA;
  float* nxt = trgB;

  for (int l = 0; l < Lc; ++l) {
    for (int att = 0; att < 2; ++att) {            // 0 = self (causal), 1 = cross
      const u16*   WT = wAttnT + (((long)l * 2 + att) * 4) * Hc * Hc;
      const float* Bb = attn_b + (((long)l * 2 + att) * 4) * Hc;
      const u16*   kv = (att == 0) ? trg_bf : enc_bf;

      gemm_wmma<M_QKV><<<grd(MT, Hc, 1), blk, 0, stream>>>(
          trg_bf, WT + 0L * Hc * Hc, nullptr, q_bf, Bb + 0 * Hc, nullptr,
          Hc, Hc, Hc, 0, 0.f, 0, 0, 0);
      gemm_wmma<M_QKV><<<grd(MT, Hc, 1), blk, 0, stream>>>(
          kv, WT + 1L * Hc * Hc, nullptr, k_bf, Bb + 1 * Hc, nullptr,
          Hc, Hc, Hc, 0, 0.f, 0, 0, 0);
      gemm_wmma<M_VT><<<grd(MT, Hc, 1), blk, 0, stream>>>(
          kv, WT + 2L * Hc * Hc, nullptr, vT_bf, Bb + 2 * Hc, nullptr,
          Hc, Hc, Hc, 0, 0.f, 0, 0, 0);

      // scores = Q @ K^T / sqrt(H)   (batched over B*NH)
      gemm_wmma<M_SCORES><<<grd(SQc, SKc, Bc * NHc), blk, 0, stream>>>(
          q_bf, k_bf, scores, nullptr, nullptr, nullptr, HDc, HDc, HDc, SKc,
          1.0f / SCALEc, (long)SQc * HDc, (long)SKc * HDc, (long)SQc * SKc);

      if (att == 0) {
        softmax_kernel<0><<<dim3(SQc, Bc * NHc), blk, 0, stream>>>(
            scores, dec_mask, probs_bf, nullptr);
      } else {
        softmax_kernel<1><<<dim3(SQc, Bc * NHc), blk, 0, stream>>>(
            scores, enc_mask, probs_bf, (l == Lc - 1) ? attn_out : nullptr);
      }

      // attn @ V  (B operand = head-transposed V), heads merged into tmp_bf
      gemm_wmma<M_MERGE><<<grd(SQc, HDc, Bc * NHc), blk, 0, stream>>>(
          probs_bf, vT_bf, nullptr, tmp_bf, nullptr, nullptr, SKc, SKc, SQc, 0,
          0.f, (long)SQc * SKc, (long)HDc * SQc, 0);

      // output projection -> delta (fp32)
      gemm_wmma<M_PLAINF><<<grd(MT, Hc, 1), blk, 0, stream>>>(
          tmp_bf, WT + 3L * Hc * Hc, delta, nullptr, Bb + 3 * Hc, nullptr,
          Hc, Hc, Hc, Hc, 0.f, 0, 0, 0);

      add_ln_kernel<<<dim3((unsigned)MT), blk, 0, stream>>>(
          cur, delta, ln_g + ((long)l * 3 + att) * Hc, ln_b + ((long)l * 3 + att) * Hc,
          nxt, trg_bf);
      float* t2 = cur; cur = nxt; nxt = t2;
    }

    // FFN
    gemm_wmma<M_RELUB><<<grd(MT, PFc, 1), blk, 0, stream>>>(
        trg_bf, wFF1T + (long)l * PFc * Hc, nullptr, ffnmid_bf,
        ff_b1 + (long)l * PFc, nullptr, Hc, Hc, Hc, PFc, 0.f, 0, 0, 0);
    gemm_wmma<M_PLAINF><<<grd(MT, Hc, 1), blk, 0, stream>>>(
        ffnmid_bf, wFF2T + (long)l * Hc * PFc, delta, nullptr,
        ff_b2 + (long)l * Hc, nullptr, PFc, PFc, PFc, Hc, 0.f, 0, 0, 0);
    add_ln_kernel<<<dim3((unsigned)MT), blk, 0, stream>>>(
        cur, delta, ln_g + ((long)l * 3 + 2) * Hc, ln_b + ((long)l * 3 + 2) * Hc,
        nxt, trg_bf);
    float* t2 = cur; cur = nxt; nxt = t2;
  }

  // final projection -> d_out
  gemm_wmma<M_PLAINF><<<grd(MT, INc, 1), blk, 0, stream>>>(
      trg_bf, wOutT, out, nullptr, out_b, nullptr, Hc, Hc, Hc, INc, 0.f, 0, 0, 0);
}
